// Net_83700322665022
// MI455X (gfx1250) — matrix operand order repaired and checked by reading, not compiled, not run on hardware
//
#include <hip/hip_runtime.h>
#include <stdint.h>

// ---------------------------------------------------------------------------
// SNN forward on MI455X: cur1 = x@W1^T+b1 once; 10x { LIF1 elementwise;
// mem2 <- LIF2(mem1@W2^T + b2) }; emit (spk2, mem2).
//
// GEMMs: v_wmma_f32_16x16x32_bf16 (bf16 A/B, f32 accum).  Both GEMMs are NT
// (K contiguous in both operands):
//   - A fragments loaded direct from global (per-lane contiguous 16B chunks,
//     tracked by LOADcnt)
//   - B (weight) tiles staged block-wide into LDS with
//     global_load_async_to_lds_b128, double-buffered, tracked by ASYNCcnt —
//     kills the NW/MW-way redundant global fetch of shared weight tiles and
//     overlaps with A loads on an independent counter.
// Working set (~190MB ws) is L2-resident (192MB) across the 10-step loop.
// ---------------------------------------------------------------------------

typedef __attribute__((ext_vector_type(16))) __bf16 bf16x16;
typedef __attribute__((ext_vector_type(8)))  __bf16 bf16x8;
typedef __attribute__((ext_vector_type(4)))  __bf16 bf16x4;
typedef __attribute__((ext_vector_type(8)))  float  f32x8;
typedef __attribute__((ext_vector_type(4)))  float  f32x4;

// B tile row stride in LDS: 32 K-elems (64B) + 16B pad = 80B = 40 bf16.
// ds_load_b128 fragment reads then hit 16 distinct 4-bank groups (stride
// 5 groups mod 16, gcd(5,16)=1) -> conflict-free.
#define BROW 40

__device__ __forceinline__ void async_b128_to_lds(const __bf16* g, const __bf16* l) {
  uint32_t loff = (uint32_t)(uintptr_t)l;   // flat-shared low 32 bits = LDS byte offset
  asm volatile("global_load_async_to_lds_b128 %0, %1, off"
               :: "v"(loff), "v"((uint64_t)(uintptr_t)g) : "memory");
}

template <int N> __device__ __forceinline__ void wait_asynccnt() {
  if constexpr (N == 0)      asm volatile("s_wait_asynccnt 0x0" ::: "memory");
  else if constexpr (N == 1) asm volatile("s_wait_asynccnt 0x1" ::: "memory");
  else if constexpr (N == 2) asm volatile("s_wait_asynccnt 0x2" ::: "memory");
  else                       asm volatile("s_wait_asynccnt 0x4" ::: "memory");
}

// ---- f32 -> bf16 bulk convert (vectorized x4) ------------------------------
__global__ void f32_to_bf16_kernel(const float* __restrict__ in,
                                   __bf16* __restrict__ out, int n4) {
  int i = blockIdx.x * blockDim.x + threadIdx.x;
  if (i < n4) {
    f32x4 v = ((const f32x4*)in)[i];
    bf16x4 b;
    b[0] = (__bf16)v[0]; b[1] = (__bf16)v[1];
    b[2] = (__bf16)v[2]; b[3] = (__bf16)v[3];
    ((bf16x4*)out)[i] = b;
  }
}

// ---- LIF update on mem1 (16M elems), also emits bf16 shadow for GEMM2 ------
__global__ void lif1_kernel(const float* __restrict__ cur1,
                            float* __restrict__ mem1,
                            __bf16* __restrict__ mem1b, int n4) {
  int i = blockIdx.x * blockDim.x + threadIdx.x;
  if (i < n4) {
    f32x4 m = ((const f32x4*)mem1)[i];
    f32x4 c = ((const f32x4*)cur1)[i];
    f32x4 nm;
    bf16x4 b;
#pragma unroll
    for (int e = 0; e < 4; ++e) {
      float r = (m[e] > 15.0f) ? 15.0f : 0.0f;   // reset from PREVIOUS mem
      nm[e] = 0.5f * m[e] + c[e] - r;
      b[e]  = (__bf16)nm[e];
    }
    ((f32x4*)mem1)[i]  = nm;
    ((bf16x4*)mem1b)[i] = b;
  }
}

// ---- NT GEMM: C[M,N] = A[M,K] @ B[N,K]^T + bias, bf16 in / f32 accum -------
// Wave grid MW x NW; per-wave tile (TM*16) x (TN*16).  B tile (block-wide,
// NW*TN*16 rows x 32 K) staged in LDS via async copy, double buffered.
// If mem2 != nullptr, fuse the LIF2 membrane update instead of storing C.
template <int MW, int NW, int TM, int TN>
__global__ __launch_bounds__(MW * NW * 32)
void gemm_nt_kernel(const __bf16* __restrict__ A, const __bf16* __restrict__ B,
                    const float* __restrict__ bias, float* __restrict__ C,
                    float* __restrict__ mem2, int M, int N, int K,
                    float beta, float thr) {
  constexpr int THREADS = MW * NW * 32;
  constexpr int ROWS = NW * TN * 16;        // B rows per block tile
  constexpr int CPT  = (ROWS * 4) / THREADS; // 16B chunks per thread per tile

  __shared__ __bf16 smemB[2][ROWS * BROW];

  const int tid  = threadIdx.x;
  const int lane = tid & 31;
  const int wave = tid >> 5;
  const int lr = lane & 15;   // row/col within 16
  const int lh = lane >> 4;   // half-wave select
  const int wm = wave / NW;
  const int wn = wave % NW;

  const int mBase  = blockIdx.y * (MW * TM * 16) + wm * (TM * 16);
  const int nBase0 = blockIdx.x * ROWS;            // block B-tile origin
  const int nBase  = nBase0 + wn * (TN * 16);      // this wave's N origin

  const __bf16* Brow0 = B + (size_t)nBase0 * K;

  f32x8 acc[TM][TN];
#pragma unroll
  for (int i = 0; i < TM; ++i)
#pragma unroll
    for (int j = 0; j < TN; ++j) acc[i][j] = (f32x8)0.0f;

  const __bf16* aRow[TM];
#pragma unroll
  for (int i = 0; i < TM; ++i)
    aRow[i] = A + (size_t)(mBase + i * 16 + lr) * K;

  // cooperative async issue of one 32-deep B tile into smemB[buf]
  auto issueB = [&](int buf, int kb) {
#pragma unroll
    for (int i = 0; i < CPT; ++i) {
      const int c   = tid + i * THREADS;
      const int row = c >> 2;
      const int sub = c & 3;
      async_b128_to_lds(Brow0 + (size_t)row * K + kb + sub * 8,
                        &smemB[buf][row * BROW + sub * 8]);
    }
  };

  const int nK = K / 32;
  issueB(0, 0);

  for (int kbi = 0; kbi < nK; ++kbi) {
    const int kb  = kbi * 32;
    const int buf = kbi & 1;
    if (kbi + 1 < nK) {
      issueB(buf ^ 1, kb + 32);
      wait_asynccnt<CPT>();   // current tile's CPT ops landed (in-order)
    } else {
      wait_asynccnt<0>();
    }
    __syncthreads();          // all waves' async copies visible

    bf16x16 af[TM], bfr[TN];
#pragma unroll
    for (int i = 0; i < TM; ++i) {
      // A lane layout: elems 0..7 -> K = kb + lh*8 ..; elems 8..15 -> +16
      bf16x8 lo = *(const bf16x8*)(aRow[i] + kb + lh * 8);
      bf16x8 hi = *(const bf16x8*)(aRow[i] + kb + 16 + lh * 8);
      af[i] = __builtin_shufflevector(lo, hi, 0, 1, 2, 3, 4, 5, 6, 7,
                                      8, 9, 10, 11, 12, 13, 14, 15);
    }
#pragma unroll
    for (int j = 0; j < TN; ++j) {
      // B lane layout: elems 0..7 -> K = kb + lh*8 ..; elems 8..15 -> +16 (col = lr)
      const __bf16* p = &smemB[buf][(wn * TN * 16 + j * 16 + lr) * BROW + lh * 8];
      bf16x8 lo = *(const bf16x8*)p;
      bf16x8 hi = *(const bf16x8*)(p + 16);
      bfr[j] = __builtin_shufflevector(lo, hi, 0, 1, 2, 3, 4, 5, 6, 7,
                                       8, 9, 10, 11, 12, 13, 14, 15);
    }
#pragma unroll
    for (int i = 0; i < TM; ++i)
#pragma unroll
      for (int j = 0; j < TN; ++j)
        acc[i][j] = __builtin_amdgcn_wmma_f32_16x16x32_bf16(
            false, af[i], false, bfr[j], (short)0, acc[i][j], false, false);

    __syncthreads();          // protect buf before next-iteration overwrite
  }

  // Epilogue: C/D layout -> element e is row (lh*8 + e), col lr of the tile.
#pragma unroll
  for (int i = 0; i < TM; ++i)
#pragma unroll
    for (int j = 0; j < TN; ++j) {
      const int gn = nBase + j * 16 + lr;
      const float bv = bias[gn];
#pragma unroll
      for (int e = 0; e < 8; ++e) {
        const int gm = mBase + i * 16 + lh * 8 + e;
        const size_t idx = (size_t)gm * N + gn;
        const float cur = acc[i][j][e] + bv;
        if (mem2 != nullptr) {
          const float m2 = mem2[idx];
          const float r  = (m2 > thr) ? thr : 0.0f;
          mem2[idx] = beta * m2 + cur - r;
        } else {
          C[idx] = cur;
        }
      }
    }
}

// ---- finalize: out = [spk2, mem2] ------------------------------------------
__global__ void finalize_kernel(const float* __restrict__ mem2,
                                float* __restrict__ out, int n) {
  int i = blockIdx.x * blockDim.x + threadIdx.x;
  if (i < n) {
    const float m = mem2[i];
    out[i]     = (m > 10.0f) ? 1.0f : 0.0f;  // heaviside(mem2 - 10)
    out[n + i] = m;
  }
}

extern "C" void kernel_launch(void* const* d_in, const int* in_sizes, int n_in,
                              void* d_out, int out_size, void* d_ws,
                              size_t ws_size, hipStream_t stream) {
  (void)in_sizes; (void)n_in; (void)out_size; (void)ws_size;
  const float* x  = (const float*)d_in[0];  // [4096,1024]
  const float* W1 = (const float*)d_in[1];  // [4096,1024]
  const float* b1 = (const float*)d_in[2];  // [4096]
  const float* W2 = (const float*)d_in[3];  // [512,4096]
  const float* b2 = (const float*)d_in[4];  // [512]

  const int B = 4096, NI = 1024, NH = 4096, NO = 512;
  const size_t MB = 1024 * 1024;

  char* ws = (char*)d_ws;                      // layout (1MB-aligned):
  __bf16* xb    = (__bf16*)(ws);               //   0..8MB   x   bf16
  __bf16* w1b   = (__bf16*)(ws + 8 * MB);      //   8..16MB  W1  bf16
  __bf16* w2b   = (__bf16*)(ws + 16 * MB);     //  16..20MB  W2  bf16
  float*  cur1  = (float*) (ws + 20 * MB);     //  20..84MB  cur1 f32
  float*  mem1  = (float*) (ws + 84 * MB);     //  84..148MB mem1 f32
  __bf16* mem1b = (__bf16*)(ws + 148 * MB);    // 148..180MB mem1 bf16
  float*  mem2  = (float*) (ws + 180 * MB);    // 180..188MB mem2 f32

  // bf16 copies of GEMM operands
  {
    int n4 = (B * NI) / 4;
    f32_to_bf16_kernel<<<(n4 + 255) / 256, 256, 0, stream>>>(x, xb, n4);
    f32_to_bf16_kernel<<<(n4 + 255) / 256, 256, 0, stream>>>(W1, w1b, n4);
    int m4 = (NO * NH) / 4;
    f32_to_bf16_kernel<<<(m4 + 255) / 256, 256, 0, stream>>>(W2, w2b, m4);
  }

  // zero-init membrane potentials (graph-capturable async memsets)
  hipMemsetAsync(mem1, 0, (size_t)B * NH * sizeof(float), stream);
  hipMemsetAsync(mem2, 0, (size_t)B * NO * sizeof(float), stream);

  // GEMM1: cur1[B,NH] = x @ W1^T + b1.  8 waves, 64x64/wave, block 128x256.
  // grid = (NH/256, B/128) = (16, 32)
  gemm_nt_kernel<2, 4, 4, 4><<<dim3(NH / 256, B / 128), 256, 0, stream>>>(
      xb, w1b, b1, cur1, nullptr, B, NH, NI, 0.0f, 0.0f);

  // 10 inner steps: LIF1 elementwise, then GEMM2 with fused LIF2 epilogue.
  for (int t = 0; t < 10; ++t) {
    int n4 = (B * NH) / 4;
    lif1_kernel<<<(n4 + 255) / 256, 256, 0, stream>>>(cur1, mem1, mem1b, n4);
    // GEMM2: mem2 <- LIF(0.5, 10) of (mem1 @ W2^T + b2).  4 waves,
    // 64x32/wave, block 128x64.  grid = (NO/64, B/128) = (8, 32)
    gemm_nt_kernel<2, 2, 4, 2><<<dim3(NO / 64, B / 128), 128, 0, stream>>>(
        mem1b, w2b, b2, nullptr, mem2, B, NO, NH, 0.5f, 10.0f);
  }

  // out = [spk2 (2M), mem2 (2M)]
  {
    int n = B * NO;
    finalize_kernel<<<(n + 255) / 256, 256, 0, stream>>>(mem2, (float*)d_out, n);
  }
}
